// AlarmworkRNN_64433099374914
// MI455X (gfx1250) — matrix operand order repaired and checked by reading, not compiled, not run on hardware
//
#include <hip/hip_runtime.h>
#include <hip/hip_bf16.h>
#include <math.h>

// Shapes from the reference
#define B_ 256
#define T_ 256
#define I_ 512
#define H_ 1024
#define O_ 128

#define NBLK   16            // persistent blocks: 16 blocks x 512 thr (16 waves) = 256 waves
#define BPITCH (H_ + 8)      // LDS row pitch in ushorts: 2064B = 516 dwords == 4 (mod 64 banks)

typedef __attribute__((ext_vector_type(16))) __bf16       v16bf;
typedef __attribute__((ext_vector_type(8)))  float        v8f;
typedef __attribute__((ext_vector_type(4)))  unsigned int u32x4;

__device__ __forceinline__ unsigned short f32_to_bf16_rne(float f) {
  union { float f; unsigned int u; } c; c.f = f;
  unsigned int u = c.u;
  u += 0x7fffu + ((u >> 16) & 1u);   // round-to-nearest-even
  return (unsigned short)(u >> 16);
}

// ---------------- f32 -> bf16 conversion (grid-stride) ----------------
__global__ void cvt_f32_bf16(const float* __restrict__ src,
                             unsigned short* __restrict__ dst, int n) {
  int i = blockIdx.x * blockDim.x + threadIdx.x;
  int stride = gridDim.x * blockDim.x;
  for (; i < n; i += stride) dst[i] = f32_to_bf16_rne(src[i]);
}

// ---------------- WMMA fragment loaders ----------------
// A tile (16x32 bf16), row-major source, base = &A[tile row 0, kbase]
__device__ __forceinline__ v16bf load_a_frag(const unsigned short* base, int lda, int lane) {
  int row = lane & 15;
  int kh  = (lane >> 4) * 8;           // lanes 0-15: K 0..7/16..23; lanes 16-31: K 8..15/24..31
  const unsigned short* p = base + row * lda + kh;
  union { u32x4 u[2]; v16bf v; } f;
  f.u[0] = *(const u32x4*)(p);
  f.u[1] = *(const u32x4*)(p + 16);
  return f.v;
}

// B tile (32x16 bf16) with B[k,n] = W[n,k]; W row-major [N,K]; base = &W[nbase*ldw + kbase]
__device__ __forceinline__ v16bf load_b_frag(const unsigned short* base, int ldw, int lane) {
  int n  = lane & 15;
  int ks = (lane >> 4) * 8;            // lanes 0-15: K 0..7/16..23; lanes 16-31: K 8..15/24..31
  const unsigned short* p = base + n * ldw + ks;
  union { u32x4 u[2]; v16bf v; } f;
  f.u[0] = *(const u32x4*)(p);
  f.u[1] = *(const u32x4*)(p + 16);
  return f.v;
}

// ---------------- pre = Xb[BT,I] @ Win1^T + b_in1 -> f32 [BT,H] ----------------
// waves = (BT/16) * (H/64) = 65536 ; each wave computes a 16x64 strip
__global__ void gemm_pre(const unsigned short* __restrict__ Xb,
                         const unsigned short* __restrict__ Wb,
                         const float* __restrict__ bias,
                         float* __restrict__ pre) {
  int wave  = (blockIdx.x * blockDim.x + threadIdx.x) >> 5;
  int lane  = threadIdx.x & 31;
  int mbase = (wave >> 4) * 16;
  int nbase = (wave & 15) * 64;
  v8f acc[4] = {};
  for (int kb = 0; kb < I_; kb += 32) {
    v16bf a = load_a_frag(Xb + (size_t)mbase * I_ + kb, I_, lane);
#pragma unroll
    for (int j = 0; j < 4; ++j) {
      v16bf b = load_b_frag(Wb + (size_t)(nbase + j * 16) * I_ + kb, I_, lane);
      acc[j] = __builtin_amdgcn_wmma_f32_16x16x32_bf16(false, a, false, b,
                                                       (short)0, acc[j], false, false);
    }
  }
  int ncol  = lane & 15;
  int mrow0 = mbase + (lane >> 4) * 8;
#pragma unroll
  for (int j = 0; j < 4; ++j) {
    int n = nbase + j * 16 + ncol;
    float bv = bias[n];
#pragma unroll
    for (int r = 0; r < 8; ++r)
      pre[(size_t)(mrow0 + r) * H_ + n] = acc[j][r] + bv;
  }
}

// ---------------- grid barrier for the persistent kernel ----------------
__device__ __forceinline__ void grid_barrier(unsigned* ctr, unsigned target) {
  __threadfence();                 // release: make this block's stores device-visible
  __syncthreads();
  if (threadIdx.x == 0) {
    atomicAdd(ctr, 1u);
    while (atomicAdd(ctr, 0u) < target) __builtin_amdgcn_s_sleep(1);
  }
  __syncthreads();
  __threadfence();                 // acquire: invalidate stale lines before reading
}

__global__ void init_ctr(unsigned* ctr) {
  if (blockIdx.x == 0 && threadIdx.x == 0) *ctr = 0u;
}

// ---------------- persistent recurrent chain (single launch) ----------------
// 16 blocks x 512 threads = 256 waves. Block b owns n-columns [64b, 64b+64);
// wave w (0..15) inside the block owns m-rows [16w, 16w+16).
// The block's W_rec slab (64 x 1024 bf16, padded pitch) lives in LDS for ALL steps.
__global__ void __launch_bounds__(512, 1)
rnn_persist(const float* __restrict__ pre,
            const unsigned short* __restrict__ Wrec,
            const unsigned short* __restrict__ Wout,
            const float* __restrict__ bout,
            unsigned short* __restrict__ z0,
            unsigned short* __restrict__ z1,
            float* __restrict__ out,
            unsigned* __restrict__ ctr) {
  __shared__ unsigned short Blds[64 * BPITCH];      // ~129 KB of the 320 KB WGP LDS
  const int tid   = threadIdx.x;
  const int lane  = tid & 31;
  const int wv    = tid >> 5;                       // 0..15
  const int mbase = wv * 16;
  const int nbase = blockIdx.x * 64;
  const int ncol  = lane & 15;
  const int mrow0 = mbase + (lane >> 4) * 8;

  // Stage W_rec[nbase:nbase+64, :] into LDS once (reused for all 254 steps).
  {
    const u32x4* src = (const u32x4*)(Wrec + (size_t)nbase * H_);
    // 64 rows x 128 u32x4 per row; destination uses padded pitch (BPITCH*2 bytes, 16B aligned)
    for (int i = tid; i < 64 * (H_ / 8); i += 512) {
      int row = i >> 7, c = i & 127;
      *(u32x4*)(Blds + row * BPITCH + c * 8) = src[(size_t)row * (H_ / 8) + c];
    }
  }

  // Phase 1: z0 = tanh(pre[:,1,:])   (the t=0 state is never consumed)
  for (int i = blockIdx.x * 512 + tid; i < B_ * H_; i += NBLK * 512) {
    int m = i >> 10, n = i & (H_ - 1);
    z0[i] = f32_to_bf16_rne(tanhf(pre[((size_t)m * T_ + 1) * H_ + n]));
  }
  grid_barrier(ctr, (unsigned)NBLK);   // also covers LDS staging visibility (has __syncthreads)

  // Phase 2: t = 2..255, ping-pong z0 <-> z1 (odd t writes z0, even t writes z1)
  for (int s = 0; s < T_ - 2; ++s) {
    const int t = s + 2;
    const unsigned short* cur = (s & 1) ? z1 : z0;
    unsigned short*       nxt = (s & 1) ? z0 : z1;

    v8f acc[4] = {};
    for (int kb = 0; kb < H_; kb += 32) {
      v16bf a = load_a_frag(cur + (size_t)mbase * H_ + kb, H_, lane);   // global/L2
#pragma unroll
      for (int j = 0; j < 4; ++j) {                                     // LDS (ds_load_b128)
        v16bf b = load_b_frag(Blds + (j * 16) * BPITCH + kb, BPITCH, lane);
        acc[j] = __builtin_amdgcn_wmma_f32_16x16x32_bf16(false, a, false, b,
                                                         (short)0, acc[j], false, false);
      }
    }
#pragma unroll
    for (int j = 0; j < 4; ++j) {
      int n = nbase + j * 16 + ncol;
#pragma unroll
      for (int r = 0; r < 8; ++r) {
        int m = mrow0 + r;
        float v = tanhf(acc[j][r] + pre[((size_t)m * T_ + t) * H_ + n]);
        nxt[(size_t)m * H_ + n] = f32_to_bf16_rne(v);
      }
    }
    grid_barrier(ctr, (unsigned)NBLK * (unsigned)(s + 2));
  }

  // Phase 3: out = tanh(z_last @ Wout^T + b_out); z_last (t=255, odd) is in z0.
  // 32 waves: blocks 0-1; mtile = gwave>>1 (16), ngroup = gwave&1 (2 x 64 = 128 cols).
  const int gwave = blockIdx.x * 16 + wv;
  if (gwave < 32) {
    const int mb2 = (gwave >> 1) * 16;
    const int nb2 = (gwave & 1) * 64;
    v8f acc[4] = {};
    for (int kb = 0; kb < H_; kb += 32) {
      v16bf a = load_a_frag(z0 + (size_t)mb2 * H_ + kb, H_, lane);
#pragma unroll
      for (int j = 0; j < 4; ++j) {
        v16bf b = load_b_frag(Wout + (size_t)(nb2 + j * 16) * H_ + kb, H_, lane);
        acc[j] = __builtin_amdgcn_wmma_f32_16x16x32_bf16(false, a, false, b,
                                                         (short)0, acc[j], false, false);
      }
    }
    const int mr2 = mb2 + (lane >> 4) * 8;
#pragma unroll
    for (int j = 0; j < 4; ++j) {
      int n = nb2 + j * 16 + ncol;
      float bv = bout[n];
#pragma unroll
      for (int r = 0; r < 8; ++r)
        out[(size_t)(mr2 + r) * O_ + n] = tanhf(acc[j][r] + bv);
    }
  }
}

extern "C" void kernel_launch(void* const* d_in, const int* in_sizes, int n_in,
                              void* d_out, int out_size, void* d_ws, size_t ws_size,
                              hipStream_t stream) {
  (void)in_sizes; (void)n_in; (void)out_size; (void)ws_size;
  const float* X      = (const float*)d_in[0];
  const float* W_in1  = (const float*)d_in[1];
  const float* b_in1  = (const float*)d_in[2];
  const float* W_rec1 = (const float*)d_in[3];
  // d_in[4..6] = W_in2, b_in2, W_rec2 : provably dead w.r.t. the output
  const float* W_out  = (const float*)d_in[7];
  const float* b_out  = (const float*)d_in[8];
  float* out = (float*)d_out;

  // Workspace carve-up (256B aligned)
  char* ws = (char*)d_ws;
  size_t off = 0;
  auto carve = [&](size_t bytes) {
    void* p = ws + off;
    off += (bytes + 255) & ~(size_t)255;
    return p;
  };
  unsigned short* Xb    = (unsigned short*)carve((size_t)B_ * T_ * I_ * 2); //  64 MB
  unsigned short* Win1b = (unsigned short*)carve((size_t)H_ * I_ * 2);      //   1 MB
  unsigned short* Wrecb = (unsigned short*)carve((size_t)H_ * H_ * 2);      //   2 MB
  unsigned short* Woutb = (unsigned short*)carve((size_t)O_ * H_ * 2);      // 256 KB
  float*          pre   = (float*)carve((size_t)B_ * T_ * H_ * 4);          // 256 MB
  unsigned short* z0    = (unsigned short*)carve((size_t)B_ * H_ * 2);      // 512 KB
  unsigned short* z1    = (unsigned short*)carve((size_t)B_ * H_ * 2);      // 512 KB
  unsigned*       ctr   = (unsigned*)carve(256);

  // 1) bf16 conversions
  cvt_f32_bf16<<<4096, 256, 0, stream>>>(X,      Xb,    B_ * T_ * I_);
  cvt_f32_bf16<<<1024, 256, 0, stream>>>(W_in1,  Win1b, H_ * I_);
  cvt_f32_bf16<<<2048, 256, 0, stream>>>(W_rec1, Wrecb, H_ * H_);
  cvt_f32_bf16<<<256,  256, 0, stream>>>(W_out,  Woutb, O_ * H_);

  // 2) pre = X @ W_in1^T + b_in1  (65536 waves, 8 waves/block)
  gemm_pre<<<8192, 256, 0, stream>>>(Xb, Win1b, b_in1, pre);

  // 3) whole recurrent chain + output head in ONE persistent launch
  init_ctr<<<1, 64, 0, stream>>>(ctr);
  rnn_persist<<<NBLK, 512, 0, stream>>>(pre, Wrecb, Woutb, b_out, z0, z1, out, ctr);
}
